// MMDLNetV0_4896262717549
// MI455X (gfx1250) — matrix operand order repaired and checked by reading, not compiled, not run on hardware
//
#include <hip/hip_runtime.h>
#include <hip/hip_bf16.h>

// ===========================================================================
// MMDLNet forward for gfx1250 (MI455X).
// All GEMM-shaped math: v_wmma_f32_16x16x32_bf16, wave32.
// Hot paths use bf16 operands -> each fragment is two 16B b128 loads.
// A-fragments are hoisted/reused across output tiles (strip-mined waves with
// 8 accumulators) so the dominant loops approach 2 loads + 1 wmma per tile.
//
// Exact gconv simplification: broadcast input =>
//   h[b,i,c] = relu( (e @ W)[b,i] * colsum(adj)[c] ).
//
// Input flattening (JAX pytree = sorted dict keys):
//  0:A 1:B 2:C 3:Y_true
//  4..11 att_a{bk,bo,bq,bv,wk,wo,wq,wv} 12..19 att_b 20..27 att_c 28..35 att_s
//  36 fsn.b 37 fsn.w  38 adj 39 wa 40 wb 41 wc 42 ws
//  43 last.b 44 last.w 45 mask  46/47 pre_a.b/w 48/49 pre_b.b/w 50/51 pre_c.b/w
// ===========================================================================

#define BATCH 32
#define HID   1024
#define NC    128
#define BOT   32
#define SEQ   1056
#define SEQT  66
#define LDSC  1072          // f32 score row stride in LDS

typedef unsigned short u16b;
typedef __attribute__((ext_vector_type(16))) __bf16       v16bf;
typedef __attribute__((ext_vector_type(8)))  float        v8f;
typedef __attribute__((ext_vector_type(8)))  unsigned int v8u;
typedef __attribute__((ext_vector_type(4)))  unsigned int v4u;

union FragU { v8u u; v16bf bf; };

// ---- scalar conversions ----------------------------------------------------
__device__ __forceinline__ unsigned int f2bf(float x) {
  unsigned int u = __float_as_uint(x);
  return (u + 0x7FFFu + ((u >> 16) & 1u)) >> 16;
}
__device__ __forceinline__ unsigned int pack2(float lo, float hi) {
  return (f2bf(lo) & 0xFFFFu) | (f2bf(hi) << 16);
}

__device__ __forceinline__ v8f wmma32(v16bf a, v16bf b, v8f c) {
  return __builtin_amdgcn_wmma_f32_16x16x32_bf16(false, a, false, b,
                                                 (short)0, c, false, false);
}

// ---- fast bf16 fragment loaders (two b128 loads each) ----------------------
__device__ __forceinline__ v16bf load_a_bf(const u16b* A, int ldA,
                                           int m0, int k0, int lane) {
  const int m  = m0 + (lane & 15);
  const int kb = (lane >> 4) * 8;
  const v4u* p = (const v4u*)(A + (size_t)m * ldA + k0 + kb);
  const v4u lo = p[0];        // k = kb+0..7
  const v4u hi = p[2];        // k = 16+kb+0..7
  FragU f;
  f.u[0]=lo[0]; f.u[1]=lo[1]; f.u[2]=lo[2]; f.u[3]=lo[3];
  f.u[4]=hi[0]; f.u[5]=hi[1]; f.u[6]=hi[2]; f.u[7]=hi[3];
  return f.bf;
}
__device__ __forceinline__ v16bf load_b_bf(const u16b* BT, int ldB,
                                           int k0, int n0, int lane) {
  const int n  = n0 + (lane & 15);
  const int kb = (lane >> 4) * 8;
  const v4u* p = (const v4u*)(BT + (size_t)n * ldB + k0 + kb);
  const v4u lo = p[0];
  const v4u hi = p[2];
  FragU f;
  f.u[0]=lo[0]; f.u[1]=lo[1]; f.u[2]=lo[2]; f.u[3]=lo[3];
  f.u[4]=hi[0]; f.u[5]=hi[1]; f.u[6]=hi[2]; f.u[7]=hi[3];
  return f.bf;
}

// ---- slow f32 fragment loaders (strided, convert on load; cold paths) ------
__device__ __forceinline__ v16bf load_a_f32(const float* A, long sAm, long sAk,
                                            int m0, int k0, int lane) {
  const int m  = m0 + (lane & 15);
  const int kb = (lane >> 4) * 8;
  FragU f;
#pragma unroll
  for (int j = 0; j < 8; ++j) {
    const int kk = (j < 4) ? (kb + 2 * j) : (16 + kb + 2 * (j - 4));
    const long base = (long)m * sAm + (long)(k0 + kk) * sAk;
    f.u[j] = pack2(A[base], A[base + sAk]);
  }
  return f.bf;
}
__device__ __forceinline__ v16bf load_b_f32(const float* B, long sBk, long sBn,
                                            int k0, int n0, int lane) {
  const int n  = n0 + (lane & 15);
  const int kb = (lane >> 4) * 8;
  FragU f;
#pragma unroll
  for (int j = 0; j < 8; ++j) {
    const int kk = (j < 4) ? (kb + 2 * j) : (16 + kb + 2 * (j - 4));
    const long base = (long)(k0 + kk) * sBk + (long)n * sBn;
    f.u[j] = pack2(B[base], B[base + sBk]);
  }
  return f.bf;
}

__device__ __forceinline__ v8f tile_ff(const float* A, long sAm, long sAk,
                                       const float* B, long sBk, long sBn,
                                       int m0, int n0, int K, int lane) {
  v8f acc = {0.f,0.f,0.f,0.f,0.f,0.f,0.f,0.f};
  for (int k0 = 0; k0 < K; k0 += 32)
    acc = wmma32(load_a_f32(A, sAm, sAk, m0, k0, lane),
                 load_b_f32(B, sBk, sBn, k0, n0, lane), acc);
  return acc;
}

// ===========================================================================
// Kernels
// ===========================================================================

// Weight prep: W[k][n] f32 -> WT[n][k] bf16
__global__ void k_wt(const float* W, int K, int N, u16b* WT) {
  const long total = (long)K * N;
  for (long i = blockIdx.x * (long)blockDim.x + threadIdx.x; i < total;
       i += (long)gridDim.x * blockDim.x) {
    const long k = i / N, n = i - k * N;
    WT[n * K + k] = (u16b)f2bf(W[i]);
  }
}

// C[16 x 128 strip] = act(A_f32 @ WT^T + bias). One wave per strip; the f32
// A-fragment conversion is amortized over 8 WMMAs (8 f32 accumulators).
__global__ __launch_bounds__(32)
void k_gemm_fb(const float* A, int ldA, const u16b* BT, int ldB,
               const float* bias, float* C, int N, int K,
               int do_relu, int has_bias) {
  const int lane = threadIdx.x;
  const int nstrips = N >> 7;                     // 128-wide strips
  const int mt = blockIdx.x / nstrips;
  const int ns = blockIdx.x % nstrips;
  const v8f zero = {0.f,0.f,0.f,0.f,0.f,0.f,0.f,0.f};
  v8f acc[8];
#pragma unroll
  for (int t = 0; t < 8; ++t) acc[t] = zero;
  for (int k0 = 0; k0 < K; k0 += 32) {
    const v16bf a = load_a_f32(A, ldA, 1, mt * 16, k0, lane);
#pragma unroll
    for (int t = 0; t < 8; ++t)
      acc[t] = wmma32(a, load_b_bf(BT, ldB, k0, ns * 128 + t * 16, lane), acc[t]);
  }
  const int mb = mt * 16 + ((lane >> 4) << 3);
#pragma unroll
  for (int t = 0; t < 8; ++t) {
    const int n = ns * 128 + t * 16 + (lane & 15);
#pragma unroll
    for (int r = 0; r < 8; ++r) {
      float v = acc[t][r];
      if (has_bias) v += bias[n];
      if (do_relu)  v = fmaxf(v, 0.f);
      C[(long)(mb + r) * N + n] = v;
    }
  }
}

// Generic strided f32 GEMM (fsn init only). bias_mode: 0 none,1 per-n,2 per-m.
__global__ __launch_bounds__(32)
void k_gemm_ff(const float* A, long sAm, long sAk, long sAb,
               const float* B, long sBk, long sBn, long sBb,
               const float* bias, int bias_mode,
               float* C, long sCm, long sCn, long sCb,
               int N, int K) {
  const int lane = threadIdx.x;
  const int ntiles = N >> 4;
  const int mt = blockIdx.x / ntiles, nt = blockIdx.x % ntiles;
  const int b = blockIdx.y;
  v8f acc = tile_ff(A + (long)b * sAb, sAm, sAk, B + (long)b * sBb, sBk, sBn,
                    mt * 16, nt * 16, K, lane);
  const int n  = nt * 16 + (lane & 15);
  const int mb = mt * 16 + ((lane >> 4) << 3);
  float* Cb = C + (long)b * sCb;
#pragma unroll
  for (int r = 0; r < 8; ++r) {
    float v = acc[r];
    if (bias_mode == 1) v += bias[n];
    else if (bias_mode == 2) v += bias[mb + r];
    Cb[(long)(mb + r) * sCm + (long)n * sCn] = v;
  }
}

__global__ void k_colsum(const float* adj, float* s) {
  const int c = threadIdx.x;
  float acc = 0.f;
  for (int kk = 0; kk < NC; ++kk) acc += adj[kk * NC + c];
  s[c] = acc;
}

__global__ void k_mean3(const float* a, const float* b, const float* c,
                        float* o, int n) {
  int i = blockIdx.x * blockDim.x + threadIdx.x;
  if (i < n) o[i] = (a[i] + b[i] + c[i]) * (1.f / 3.f);
}

// h[b,i,c] = relu(u[b,i] * s[c])
__global__ void k_build_h(const float* u, const float* s, float* h) {
  const long total = (long)BATCH * HID * NC;
  for (long idx = blockIdx.x * (long)blockDim.x + threadIdx.x; idx < total;
       idx += (long)gridDim.x * blockDim.x) {
    h[idx] = fmaxf(u[idx >> 7] * s[idx & (NC - 1)], 0.f);
  }
}

// q/k/v projections over concat tokens [h ; fsn]. One wave per 16-token tile:
// converts the A tile once (4 fragments) and reuses it for all 8 out tiles.
// Emits q,k row-major bf16 and v TRANSPOSED bf16 (vT[b][d][key]).
__global__ __launch_bounds__(32)
void k_proj(const float* hsrc, const float* fsn,
            const u16b* wqT, const float* bq,
            const u16b* wkT, const float* bk,
            const u16b* wvT, const float* bv,
            u16b* q, u16b* k, u16b* vT) {
  const int lane = threadIdx.x;
  const int mt = blockIdx.x;                 // 0..65 token tile
  const int b  = blockIdx.y;
  const int which = blockIdx.z;
  const u16b*  WT = (which == 0) ? wqT : (which == 1) ? wkT : wvT;
  const float* bb = (which == 0) ? bq  : (which == 1) ? bk  : bv;

  const int row0 = mt * 16;
  const float* Asrc = (row0 < HID)
      ? hsrc + ((long)b * HID + row0) * NC
      : fsn  + ((long)b * BOT + (row0 - HID)) * NC;

  v16bf af[4];
#pragma unroll
  for (int c = 0; c < 4; ++c) af[c] = load_a_f32(Asrc, NC, 1, 0, c * 32, lane);

  const int mb = (lane >> 4) << 3;
  for (int nt = 0; nt < 8; ++nt) {
    v8f acc = {0.f,0.f,0.f,0.f,0.f,0.f,0.f,0.f};
#pragma unroll
    for (int c = 0; c < 4; ++c)
      acc = wmma32(af[c], load_b_bf(WT, NC, c * 32, nt * 16, lane), acc);
    const int n = nt * 16 + (lane & 15);
#pragma unroll
    for (int r = 0; r < 8; ++r) {
      const u16b val = (u16b)f2bf(acc[r] + bb[n]);
      const int tok = row0 + mb + r;
      if (which == 2) vT[((long)b * NC + n) * SEQ + tok] = val;
      else if (which == 0) q[((long)b * SEQ + tok) * NC + n] = val;
      else                 k[((long)b * SEQ + tok) * NC + n] = val;
    }
  }
}

// Exact SDPA + output projection + residual epilogue.
// One wave per (batch, 16-query tile); 16x1056 score row in LDS.
extern __shared__ float smem[];
__global__ __launch_bounds__(32)
void k_attn(const u16b* q, const u16b* k, const u16b* vT,
            const u16b* woT, const float* bo,
            float* hres, float* fout) {
  float* sc = smem;                                   // [16][LDSC] f32
  u16b*  pb = (u16b*)(smem + 16 * LDSC);              // [16][SEQ]  bf16 probs
  u16b*  ob = pb + 16 * SEQ;                          // [16][128]  bf16 O tile
  const int lane = threadIdx.x;
  const int qt = blockIdx.x;
  const int b  = blockIdx.y;
  const u16b* qb = q  + (long)b * SEQ * NC;
  const u16b* kb = k  + (long)b * SEQ * NC;
  const u16b* vb = vT + (long)b * NC * SEQ;
  const float scale = 0.08838834764831845f;           // NC^-0.5
  const int mb = (lane >> 4) << 3;
  const v8f zero = {0.f,0.f,0.f,0.f,0.f,0.f,0.f,0.f};

  // Phase 1: S = Q Kt * scale. Q A-fragments hoisted across all 66 key tiles.
  {
    v16bf aq[4];
#pragma unroll
    for (int c = 0; c < 4; ++c) aq[c] = load_a_bf(qb, NC, qt * 16, c * 32, lane);
    for (int kt = 0; kt < SEQT; ++kt) {
      v8f acc = zero;
#pragma unroll
      for (int c = 0; c < 4; ++c)
        acc = wmma32(aq[c], load_b_bf(kb, NC, c * 32, kt * 16, lane), acc);
      const int n = kt * 16 + (lane & 15);
#pragma unroll
      for (int r = 0; r < 8; ++r) sc[(mb + r) * LDSC + n] = acc[r] * scale;
    }
  }
  __syncthreads();

  // Phase 2: exact softmax per query row; emit packed-bf16 probabilities.
  if (lane < 16) {
    const float* row = sc + lane * LDSC;
    unsigned int* prow = (unsigned int*)(pb + lane * SEQ);
    float mx = -3.4e38f;
    for (int i = 0; i < SEQ; ++i) mx = fmaxf(mx, row[i]);
    float sum = 0.f;
    for (int i = 0; i < SEQ; ++i) sum += __expf(row[i] - mx);
    const float inv = 1.f / sum;
    for (int i = 0; i < SEQ; i += 2)
      prow[i >> 1] = pack2(__expf(row[i] - mx) * inv,
                           __expf(row[i + 1] - mx) * inv);
  }
  __syncthreads();

  // Phase 3: O = P @ V. One P fragment per k-chunk feeds all 8 dim tiles.
  {
    v8f acc[8];
#pragma unroll
    for (int t = 0; t < 8; ++t) acc[t] = zero;
    for (int k0 = 0; k0 < SEQ; k0 += 32) {
      const v16bf ap = load_a_bf(pb, SEQ, 0, k0, lane);
#pragma unroll
      for (int t = 0; t < 8; ++t)
        acc[t] = wmma32(ap, load_b_bf(vb, SEQ, k0, t * 16, lane), acc[t]);
    }
#pragma unroll
    for (int t = 0; t < 8; ++t) {
      const int n = t * 16 + (lane & 15);
#pragma unroll
      for (int r = 0; r < 8; ++r) ob[(mb + r) * NC + n] = (u16b)f2bf(acc[t][r]);
    }
  }
  __syncthreads();

  // Phase 4: Y = O @ Wo + bo. O A-fragments hoisted across the 8 out tiles.
  {
    v16bf ao[4];
#pragma unroll
    for (int c = 0; c < 4; ++c) ao[c] = load_a_bf(ob, NC, 0, c * 32, lane);
    for (int et = 0; et < 8; ++et) {
      v8f acc = zero;
#pragma unroll
      for (int c = 0; c < 4; ++c)
        acc = wmma32(ao[c], load_b_bf(woT, NC, c * 32, et * 16, lane), acc);
      const int e = et * 16 + (lane & 15);
#pragma unroll
      for (int r = 0; r < 8; ++r) {
        const int ntok = qt * 16 + mb + r;
        const float val = acc[r] + bo[e];
        if (ntok < HID) hres[((long)b * HID + ntok) * NC + e] += val;
        else            fout[((long)b * BOT + (ntok - HID)) * NC + e] = val;
      }
    }
  }
}

__global__ void k_fsnupd(float* fsn, const float* f1, const float* f2, int n) {
  int i = blockIdx.x * blockDim.x + threadIdx.x;
  if (i < n) fsn[i] += f1[i] + f2[i];
}

// score[b,o] = sum_c ( sum_i fsn[b,i,c]*lw[i,o] + lb[o] ) * mask[o,c]
__global__ void k_score(const float* fsn, const float* lw, const float* lb,
                        const float* mask, float* out_score) {
  const int o = threadIdx.x;
  const int b = blockIdx.x;
  float s = 0.f;
  for (int c = 0; c < NC; ++c) {
    float acc = 0.f;
#pragma unroll 4
    for (int i = 0; i < BOT; ++i)
      acc += fsn[((long)b * BOT + i) * NC + c] * lw[i * NC + o];
    s += (acc + lb[o]) * mask[o * NC + c];
  }
  out_score[b * NC + o] = s;
}

__global__ void k_loss(const float* score, const float* y, float* out) {
  __shared__ float red[256];
  const int t = threadIdx.x;
  float acc = 0.f;
  for (int i = t; i < BATCH * NC; i += 256) {
    const float s = score[i];
    acc += fmaxf(s, 0.f) - s * y[i] + log1pf(__expf(-fabsf(s)));
  }
  red[t] = acc;
  __syncthreads();
  for (int st = 128; st > 0; st >>= 1) {
    if (t < st) red[t] += red[t + st];
    __syncthreads();
  }
  if (t == 0) out[BATCH * NC] = red[0] / (float)(BATCH * NC);
}

// ===========================================================================
// Host orchestration
// ===========================================================================
extern "C" void kernel_launch(void* const* d_in, const int* in_sizes, int n_in,
                              void* d_out, int out_size, void* d_ws, size_t ws_size,
                              hipStream_t stream) {
  (void)in_sizes; (void)n_in; (void)out_size; (void)ws_size;
  auto F = [&](int i) { return (const float*)d_in[i]; };

  const float *inA = F(0), *inB = F(1), *inC = F(2), *Yt = F(3);
  struct AttW { const float *wq, *bq, *wk, *bk, *wv, *bv, *wo, *bo; };
  auto mkatt = [&](int base) {  // sorted leaves: bk bo bq bv wk wo wq wv
    AttW a; a.bk = F(base + 0); a.bo = F(base + 1); a.bq = F(base + 2);
    a.bv = F(base + 3); a.wk = F(base + 4); a.wo = F(base + 5);
    a.wq = F(base + 6); a.wv = F(base + 7); return a;
  };
  const AttW attA = mkatt(4), attB = mkatt(12), attC = mkatt(20), attS = mkatt(28);
  const float *fsnb = F(36), *fsnw = F(37);
  const float *adj = F(38), *wa = F(39), *wb = F(40), *wc = F(41), *wsw = F(42);
  const float *lastb = F(43), *lastw = F(44), *mask = F(45);
  const float *pab = F(46), *paw = F(47), *pbb = F(48), *pbw = F(49),
              *pcb = F(50), *pcw = F(51);

  // ---- workspace (~113 MB) ------------------------------------------------
  char* wsb = (char*)d_ws;
  size_t off = 0;
  auto allocf = [&](size_t nf) {
    float* p = (float*)(wsb + off);
    off += ((nf * 4 + 255) / 256) * 256; return p;
  };
  auto allocb = [&](size_t ne) {
    u16b* p = (u16b*)(wsb + off);
    off += ((ne * 2 + 255) / 256) * 256; return p;
  };
  float *ea = allocf(BATCH * HID), *eb = allocf(BATCH * HID),
        *ec = allocf(BATCH * HID), *em = allocf(BATCH * HID);
  float *ua = allocf(BATCH * HID), *ub = allocf(BATCH * HID),
        *uc = allocf(BATCH * HID), *us = allocf(BATCH * HID);
  float *svec = allocf(NC);
  float *ha = allocf((size_t)BATCH * HID * NC), *hb = allocf((size_t)BATCH * HID * NC),
        *hc = allocf((size_t)BATCH * HID * NC), *hs = allocf((size_t)BATCH * HID * NC);
  float *fsn = allocf((size_t)BATCH * BOT * NC);
  float *f1 = allocf((size_t)BATCH * BOT * NC), *f2 = allocf((size_t)BATCH * BOT * NC);
  u16b *qbf = allocb((size_t)BATCH * SEQ * NC), *kbf = allocb((size_t)BATCH * SEQ * NC),
       *vTb = allocb((size_t)BATCH * NC * SEQ);
  u16b *pawT = allocb((size_t)2048 * HID), *pbwT = allocb((size_t)2048 * HID),
       *pcwT = allocb((size_t)1024 * HID);
  u16b *waT = allocb((size_t)HID * HID), *wbT = allocb((size_t)HID * HID),
       *wcT = allocb((size_t)HID * HID), *wsT = allocb((size_t)HID * HID);
  struct AttT { u16b *wqT, *wkT, *wvT, *woT; };
  AttT tA, tB, tC, tS;
  for (AttT* t : {&tA, &tB, &tC, &tS}) {
    t->wqT = allocb(NC * NC); t->wkT = allocb(NC * NC);
    t->wvT = allocb(NC * NC); t->woT = allocb(NC * NC);
  }

  // ---- 0) weight transposition to bf16 ------------------------------------
  auto wt = [&](const float* w, int K, int N, u16b* o) {
    long tot = (long)K * N;
    unsigned g = (unsigned)((tot + 255) / 256);
    if (g > 2048) g = 2048;
    k_wt<<<dim3(g), 256, 0, stream>>>(w, K, N, o);
  };
  wt(paw, 2048, HID, pawT); wt(pbw, 2048, HID, pbwT); wt(pcw, 1024, HID, pcwT);
  wt(wa, HID, HID, waT); wt(wb, HID, HID, wbT); wt(wc, HID, HID, wcT);
  wt(wsw, HID, HID, wsT);
  const AttW* aws[4] = {&attA, &attB, &attC, &attS};
  AttT* ats[4] = {&tA, &tB, &tC, &tS};
  for (int i = 0; i < 4; ++i) {
    wt(aws[i]->wq, NC, NC, ats[i]->wqT); wt(aws[i]->wk, NC, NC, ats[i]->wkT);
    wt(aws[i]->wv, NC, NC, ats[i]->wvT); wt(aws[i]->wo, NC, NC, ats[i]->woT);
  }

  // ---- 1) pre-encoders: e = relu(X @ W + b) -------------------------------
  k_gemm_fb<<<dim3(2 * 8), 32, 0, stream>>>(inA, 2048, pawT, 2048, pab, ea, HID, 2048, 1, 1);
  k_gemm_fb<<<dim3(2 * 8), 32, 0, stream>>>(inB, 2048, pbwT, 2048, pbb, eb, HID, 2048, 1, 1);
  k_gemm_fb<<<dim3(2 * 8), 32, 0, stream>>>(inC, 1024, pcwT, 1024, pcb, ec, HID, 1024, 1, 1);
  k_mean3<<<dim3((BATCH * HID + 255) / 256), 256, 0, stream>>>(ea, eb, ec, em, BATCH * HID);

  // ---- 2) gcn: u = e @ W; h = relu(outer(u, colsum(adj))) -----------------
  k_colsum<<<1, NC, 0, stream>>>(adj, svec);
  const float* einp[4] = {ea, eb, ec, em};
  const u16b*  winp[4] = {waT, wbT, wcT, wsT};
  float* uout[4] = {ua, ub, uc, us};
  float* hout[4] = {ha, hb, hc, hs};
  for (int i = 0; i < 4; ++i) {
    k_gemm_fb<<<dim3(2 * 8), 32, 0, stream>>>(einp[i], HID, winp[i], HID,
                                              nullptr, uout[i], HID, HID, 0, 0);
    k_build_h<<<dim3(4096), 256, 0, stream>>>(uout[i], svec, hout[i]);
  }

  // ---- 3) fsn init: fsn[b,o,c] = sum_i ha[b,i,c] w[i,o] + b[o] ------------
  k_gemm_ff<<<dim3(2 * 8, BATCH), 32, 0, stream>>>(
      fsnw, 1, BOT, 0,               // A[o][i] = w[i*BOT+o]
      ha, NC, 1, (long)HID * NC,     // B[i][c] = ha[b,i,c]
      fsnb, 2,                       // bias per-row (o)
      fsn, NC, 1, (long)BOT * NC, NC, HID);

  // ---- 4) six fuse() calls ------------------------------------------------
  const int shbytes = 16 * LDSC * 4 + 16 * SEQ * 2 + 16 * NC * 2;  // 104 KB
  auto run_fuse = [&](float* h, const AttW& a, const AttT& t, float* fbuf) {
    k_proj<<<dim3(SEQT, BATCH, 3), 32, 0, stream>>>(
        h, fsn, t.wqT, a.bq, t.wkT, a.bk, t.wvT, a.bv, qbf, kbf, vTb);
    k_attn<<<dim3(SEQT, BATCH), 32, shbytes, stream>>>(
        qbf, kbf, vTb, t.woT, a.bo, h, fbuf);
  };
  const int nfsn = BATCH * BOT * NC;
  run_fuse(hs, attS, tS, f1); run_fuse(hb, attB, tB, f2);
  k_fsnupd<<<dim3((nfsn + 255) / 256), 256, 0, stream>>>(fsn, f1, f2, nfsn);
  run_fuse(hs, attS, tS, f1); run_fuse(hc, attC, tC, f2);
  k_fsnupd<<<dim3((nfsn + 255) / 256), 256, 0, stream>>>(fsn, f1, f2, nfsn);
  run_fuse(hs, attS, tS, f1); run_fuse(ha, attA, tA, f2);
  k_fsnupd<<<dim3((nfsn + 255) / 256), 256, 0, stream>>>(fsn, f1, f2, nfsn);

  // ---- 5) head ------------------------------------------------------------
  float* out = (float*)d_out;
  k_score<<<dim3(BATCH), NC, 0, stream>>>(fsn, lastw, lastb, mask, out);
  k_loss<<<1, 256, 0, stream>>>(out, Yt, out);
}
